// FPN_LSS_Mamba_FOBA_14370960572904
// MI455X (gfx1250) — matrix operand order repaired and checked by reading, not compiled, not run on hardware
//
#include <hip/hip_runtime.h>
#include <math.h>

// ---------------------------------------------------------------------------
// CDNA5 (gfx1250) forward implementation of the FPN+VSS(Mamba) reference.
// GEMM-shaped math -> v_wmma_f32_16x16x32_bf16 (bf16 in, f32 accum).
// LDS tiles in bf16 (A row-major, B transposed) -> ds_load_b128 fragments.
// Interior tiles filled with b128 global loads (uniform stride dispatch);
// edge tiles use clamped loads with multiply-mask (no exec divergence).
// Conv weight tiles staged by the Tensor Data Mover + s_wait_tensorcnt.
// ---------------------------------------------------------------------------

typedef __attribute__((ext_vector_type(16))) __bf16 v16bf;
typedef __attribute__((ext_vector_type(8)))  __bf16 v8bf;
typedef __attribute__((ext_vector_type(8)))  float  v8f;
typedef __attribute__((ext_vector_type(4)))  unsigned int v4u;
typedef __attribute__((ext_vector_type(4)))  int v4i;
typedef __attribute__((ext_vector_type(8)))  int v8i;
typedef __attribute__((ext_vector_type(4), aligned(4))) float v4fu;  // 4B-aligned

#define NST 16          // N_STATE
#define BM 64
#define BN 64
#define BK 32
#define GT 256          // threads per GEMM block (8 waves)
#define LDT (BK + 8)    // bf16 LDS row stride (80B: keeps 16B alignment)

__device__ __forceinline__ int imin(int a, int b) { return a < b ? a : b; }
__device__ __forceinline__ int imax(int a, int b) { return a > b ? a : b; }

__device__ __forceinline__ float act_apply(float v, int act) {
  switch (act) {
    case 1: return v > 0.f ? v : 0.f;                 // relu
    case 2: return v > 0.f ? v : 0.01f * v;           // leaky_relu (jax default)
    case 3: return v / (1.f + __expf(-v));            // silu
    case 4: return v > 20.f ? v : log1pf(__expf(v));  // softplus
    default: return v;
  }
}

__device__ __forceinline__ v16bf cat16(v8bf lo, v8bf hi) {
  return __builtin_shufflevector(lo, hi, 0, 1, 2, 3, 4, 5, 6, 7,
                                 8, 9, 10, 11, 12, 13, 14, 15);
}

// A fragment (ISA 7.12.2, 16-bit A 16x32): row = lane%16,
// K(e) = 16*(e/8) + 8*half + (e%8) -> two contiguous 8-elem runs.
__device__ __forceinline__ v16bf load_afrag(const __bf16* As, int row, int hl) {
  const __bf16* p = As + row * LDT;
  v8bf lo = *(const v8bf*)(p + 8 * hl);
  v8bf hi = *(const v8bf*)(p + 16 + 8 * hl);
  return cat16(lo, hi);
}

// B fragment from transposed tile Bt[n][k]: col = lane%16, K = 16*(e/8) + 8*half + (e%8).
__device__ __forceinline__ v16bf load_bfrag(const __bf16* Bt, int n, int hl) {
  const __bf16* p = Bt + n * LDT;
  v8bf lo = *(const v8bf*)(p + 8 * hl);
  v8bf hi = *(const v8bf*)(p + 16 + 8 * hl);
  return cat16(lo, hi);
}

__device__ __forceinline__ v8bf pack8(v4fu f0, v4fu f1) {
  v8bf o;
#pragma unroll
  for (int j = 0; j < 4; ++j) { o[j] = (__bf16)f0[j]; o[4 + j] = (__bf16)f1[j]; }
  return o;
}

// ------------------------- WMMA GEMM (strided) -----------------------------
// C[m,n] = act( (sum_k A[m,k]*B[k,n]) * scale[n] + bias[n] )
__global__ __launch_bounds__(GT)
void gemm_wmma_kernel(const float* __restrict__ A, long long sAm, long long sAk,
                      const float* __restrict__ B, long long sBk, long long sBn,
                      float* __restrict__ Cp, long long sCm, long long sCn,
                      int M, int N, int K,
                      const float* __restrict__ scale,
                      const float* __restrict__ bias, int act)
{
  __shared__ __attribute__((aligned(16))) __bf16 As[BM * LDT];
  __shared__ __attribute__((aligned(16))) __bf16 Bt[BN * LDT];
  const int tid  = threadIdx.x;
  const int lane = tid & 31;
  const int hl   = (lane >> 4) & 1;
  const int l16  = lane & 15;
  const int wid  = tid >> 5;
  const int wm   = wid & 1;           // 2 row-groups of 32
  const int wn   = wid >> 1;          // 4 col-groups of 16
  const int m0   = blockIdx.y * BM;
  const int n0   = blockIdx.x * BN;

  v8f acc0 = {};
  v8f acc1 = {};

  for (int k0 = 0; k0 < K; k0 += BK) {
    const bool kin  = (k0 + BK <= K);
    const bool intM = (m0 + BM <= M) && kin;
    const bool intN = (n0 + BN <= N) && kin;

    // ---------------- A tile -> As[m][k] (bf16) ----------------
    if (intM && sAk == 1) {            // row-major: b128 loads, b128 LDS store
      int r = tid >> 2, c8 = (tid & 3) * 8;
      const float* p = A + (long long)(m0 + r) * sAm + (k0 + c8);
      *(v8bf*)&As[r * LDT + c8] = pack8(*(const v4fu*)p, *(const v4fu*)(p + 4));
    } else if (intM && sAm == 1) {     // col-major: vector along m, scatter
      int k = tid >> 3, m8 = (tid & 7) * 8;
      const float* p = A + (long long)(k0 + k) * sAk + (m0 + m8);
      v4fu f0 = *(const v4fu*)p;
      v4fu f1 = *(const v4fu*)(p + 4);
#pragma unroll
      for (int j = 0; j < 4; ++j) {
        As[(m8 + j) * LDT + k]     = (__bf16)f0[j];
        As[(m8 + 4 + j) * LDT + k] = (__bf16)f1[j];
      }
    } else {                           // edge: clamped load * mask (no branch)
#pragma unroll
      for (int i = 0; i < 8; ++i) {
        int id = tid + i * GT;
        int r = id >> 5, c = id & 31;
        int m = m0 + r, k = k0 + c;
        float flt = (m < M && k < K) ? 1.f : 0.f;
        float v = A[(long long)imin(m, M - 1) * sAm + (long long)imin(k, K - 1) * sAk];
        As[r * LDT + c] = (__bf16)(v * flt);
      }
    }

    // ---------------- B tile -> Bt[n][k] (bf16, transposed) ----------------
    if (intN && sBk == 1) {            // k-contiguous: direct vector rows
      int r = tid >> 2, c8 = (tid & 3) * 8;
      const float* p = B + (long long)(n0 + r) * sBn + (k0 + c8);
      *(v8bf*)&Bt[r * LDT + c8] = pack8(*(const v4fu*)p, *(const v4fu*)(p + 4));
    } else if (intN && sBn == 1) {     // row-major: vector along n, scatter
      int k = tid >> 3, n8 = (tid & 7) * 8;
      const float* p = B + (long long)(k0 + k) * sBk + (n0 + n8);
      v4fu f0 = *(const v4fu*)p;
      v4fu f1 = *(const v4fu*)(p + 4);
#pragma unroll
      for (int j = 0; j < 4; ++j) {
        Bt[(n8 + j) * LDT + k]     = (__bf16)f0[j];
        Bt[(n8 + 4 + j) * LDT + k] = (__bf16)f1[j];
      }
    } else {
#pragma unroll
      for (int i = 0; i < 8; ++i) {
        int id = tid + i * GT;
        int n = id & 63, k = id >> 6;
        int gn = n0 + n, gk = k0 + k;
        float flt = (gn < N && gk < K) ? 1.f : 0.f;
        float v = B[(long long)imin(gk, K - 1) * sBk + (long long)imin(gn, N - 1) * sBn];
        Bt[n * LDT + k] = (__bf16)(v * flt);
      }
    }

    // prefetch next K tile while this one computes
    if (k0 + BK < K) {
      int mr = imin(m0 + (tid >> 5), M - 1);
      int kn = imin(k0 + BK + (tid & 31), K - 1);
      __builtin_prefetch(&A[(long long)mr * sAm + (long long)kn * sAk], 0, 1);
      int nn = imin(n0 + (tid & 63), N - 1);
      int k2 = imin(k0 + BK + (tid >> 6), K - 1);
      __builtin_prefetch(&B[(long long)k2 * sBk + (long long)nn * sBn], 0, 1);
    }
    __syncthreads();

    v16bf bfv = load_bfrag(Bt, wn * 16 + l16, hl);
    v16bf a0  = load_afrag(As, wm * 32 + l16, hl);
    v16bf a1  = load_afrag(As, wm * 32 + 16 + l16, hl);
    acc0 = __builtin_amdgcn_wmma_f32_16x16x32_bf16(false, a0, false, bfv,
                                                   (short)0, acc0, false, false);
    acc1 = __builtin_amdgcn_wmma_f32_16x16x32_bf16(false, a1, false, bfv,
                                                   (short)0, acc1, false, false);
    __syncthreads();
  }

  const int n = n0 + wn * 16 + l16;
#pragma unroll
  for (int t = 0; t < 2; ++t) {
    v8f acc = t ? acc1 : acc0;
#pragma unroll
    for (int r = 0; r < 8; ++r) {
      int m = m0 + wm * 32 + t * 16 + 8 * hl + r;  // D row = r + 8*half
      if (m < M && n < N) {
        float v = acc[r];
        if (scale) v *= scale[n];
        if (bias)  v += bias[n];
        Cp[(long long)m * sCm + (long long)n * sCn] = act_apply(v, act);
      }
    }
  }
}

// ------------------- implicit-GEMM 3x3 SAME conv (NCHW) ---------------------
// out[n, m] = act( scale[n]*(sum_k patch(m,k)*w[n,k]) + bias[n] ), K = Cin*9.
// Weight tile (uniform per block, row-major [Cout][K]) staged via TDM.
__global__ __launch_bounds__(GT)
void conv3x3_wmma_kernel(const float* __restrict__ in, const float* __restrict__ wgt,
                         float* __restrict__ out, int Cin, int Cout, int H, int W,
                         const float* __restrict__ scale,
                         const float* __restrict__ bias, int act)
{
  __shared__ __attribute__((aligned(16))) __bf16 As[BM * LDT];
  __shared__ __attribute__((aligned(16))) __bf16 Bt[BN * LDT];
  __shared__ __attribute__((aligned(16))) float  Bstage[BK * BN];
  const int HW = H * W;
  const int K  = Cin * 9;
  const int tid  = threadIdx.x;
  const int lane = tid & 31;
  const int hl   = (lane >> 4) & 1;
  const int l16  = lane & 15;
  const int wid  = tid >> 5;
  const int wm   = wid & 1;
  const int wn   = wid >> 1;
  const int m0   = blockIdx.y * BM;
  const int n0   = blockIdx.x * BN;

  v8f acc0 = {};
  v8f acc1 = {};

  for (int k0 = 0; k0 < K; k0 += BK) {
#if defined(__gfx1250__)
    // ---- TDM: DMA the 64x32 fp32 weight tile into LDS (zero-fill OOB) ----
    if (wid == 0) {
      unsigned ldsoff = (unsigned)(unsigned long long)(void*)Bstage;
      unsigned long long ga =
          (unsigned long long)(const void*)(wgt + (long long)n0 * K + k0);
      unsigned td0 = (unsigned)(K - k0);      // remaining elems in a row
      unsigned td1 = (unsigned)(Cout - n0);   // remaining rows
      // D# group0: count=1 | lds_addr | global_addr | type=2
      v4u g0 = { 1u, ldsoff, (unsigned)ga,
                 ((unsigned)(ga >> 32) & 0x01FFFFFFu) | (2u << 30) };
      // D# group1: data_size=4B, tensor_dim0/1, tile_dim0=32, tile_dim1=64,
      //            tensor_dim0_stride=K  (bit-packed per ISA 8.4)
      v8i g1 = { (int)(2u << 16),
                 (int)((td0 & 0xFFFFu) << 16),
                 (int)((td0 >> 16) | ((td1 & 0xFFFFu) << 16)),
                 (int)((td1 >> 16) | (32u << 16)),
                 64,
                 (int)(unsigned)K,
                 0, 0 };
      v4i gz = { 0, 0, 0, 0 };
#if __clang_major__ >= 23
      v8i gz8 = { 0, 0, 0, 0, 0, 0, 0, 0 };
      __builtin_amdgcn_tensor_load_to_lds(g0, g1, gz, gz, gz8, 0);
#else
      __builtin_amdgcn_tensor_load_to_lds(g0, g1, gz, gz, 0);
#endif
      __builtin_amdgcn_s_wait_tensorcnt(0);
    }
    __syncthreads();
    // convert staged fp32 weights -> bf16, transposed Bt[n][k]
#pragma unroll
    for (int i = 0; i < 8; ++i) {
      int id = tid + i * GT;
      int n = id >> 5, k = id & 31;
      Bt[n * LDT + k] = (__bf16)Bstage[id];
    }
#else
    // host-pass / non-gfx1250 fallback: plain clamped loads
#pragma unroll
    for (int i = 0; i < 8; ++i) {
      int id = tid + i * GT;
      int n = id & 63, k = id >> 6;
      int gn = n0 + n, gk = k0 + k;
      float flt = (gn < Cout && gk < K) ? 1.f : 0.f;
      float v = wgt[(long long)imin(gn, Cout - 1) * K + imin(gk, K - 1)];
      Bt[n * LDT + k] = (__bf16)(v * flt);
    }
#endif
    // A tile: on-the-fly im2col gather, clamped load * mask (no divergence)
#pragma unroll
    for (int i = 0; i < 8; ++i) {
      int id = tid + i * GT;
      int r = id >> 5, c = id & 31;
      int m = m0 + r, k = k0 + c;
      int kc = imin(k, K - 1);
      int ci = kc / 9, rem = kc % 9;
      int ky = rem / 3 - 1, kx = rem % 3 - 1;
      int mm = imin(m, HW - 1);
      int h = mm / W, w = mm % W;
      int ih = h + ky, iw = w + kx;
      float flt = ((m < HW) && (k < K) && (ih >= 0) && (ih < H) &&
                   (iw >= 0) && (iw < W)) ? 1.f : 0.f;
      int ihc = imin(imax(ih, 0), H - 1);
      int iwc = imin(imax(iw, 0), W - 1);
      float v = in[(long long)ci * HW + ihc * W + iwc];
      As[r * LDT + c] = (__bf16)(v * flt);
    }
    if (k0 + BK < K) {
      int nn = imin(n0 + (tid & 63), Cout - 1);
      int k2 = imin(k0 + BK + (tid >> 6), K - 1);
      __builtin_prefetch(&wgt[(long long)nn * K + k2], 0, 1);
    }
    __syncthreads();

    v16bf bfv = load_bfrag(Bt, wn * 16 + l16, hl);
    v16bf a0  = load_afrag(As, wm * 32 + l16, hl);
    v16bf a1  = load_afrag(As, wm * 32 + 16 + l16, hl);
    acc0 = __builtin_amdgcn_wmma_f32_16x16x32_bf16(false, a0, false, bfv,
                                                   (short)0, acc0, false, false);
    acc1 = __builtin_amdgcn_wmma_f32_16x16x32_bf16(false, a1, false, bfv,
                                                   (short)0, acc1, false, false);
    __syncthreads();
  }

  const int n = n0 + wn * 16 + l16;
#pragma unroll
  for (int t = 0; t < 2; ++t) {
    v8f acc = t ? acc1 : acc0;
#pragma unroll
    for (int r = 0; r < 8; ++r) {
      int m = m0 + wm * 32 + t * 16 + 8 * hl + r;
      if (m < HW && n < Cout) {
        float v = acc[r];
        if (scale) v *= scale[n];
        if (bias)  v += bias[n];
        out[(long long)n * HW + m] = act_apply(v, act);
      }
    }
  }
}

// ------------------------------ LayerNorm ----------------------------------
__global__ void ln_kernel(const float* __restrict__ x, float* __restrict__ y,
                          const float* __restrict__ g, const float* __restrict__ b,
                          int L, int C)
{
  int wid  = threadIdx.x >> 5;
  int lane = threadIdx.x & 31;
  int l = blockIdx.x * (blockDim.x >> 5) + wid;
  if (l >= L) return;
  const float* row = x + (long long)l * C;
  float s = 0.f, ss = 0.f;
  for (int c = lane; c < C; c += 32) { float v = row[c]; s += v; ss += v * v; }
#pragma unroll
  for (int o = 16; o > 0; o >>= 1) { s += __shfl_xor(s, o, 32); ss += __shfl_xor(ss, o, 32); }
  float mean = s / C;
  float var  = ss / C - mean * mean;
  float rstd = rsqrtf(var + 1e-5f);
  float* orow = y + (long long)l * C;
  for (int c = lane; c < C; c += 32)
    orow[c] = (row[c] - mean) * rstd * g[c] + b[c];
}

// ---------------- depthwise 3x3 conv (NHWC slice of hw2) + bias + silu ------
__global__ void dwconv_silu_kernel(const float* __restrict__ xi, int ldx,
                                   const float* __restrict__ w,
                                   const float* __restrict__ b,
                                   float* __restrict__ u, int H, int W, int C)
{
  long long i = blockIdx.x * (long long)blockDim.x + threadIdx.x;
  long long total = (long long)H * W * C;
  if (i >= total) return;
  int c = (int)(i % C);
  int l = (int)(i / C);
  int h = l / W, wq = l % W;
  float acc = b[c];
#pragma unroll
  for (int ky = 0; ky < 3; ++ky)
#pragma unroll
    for (int kx = 0; kx < 3; ++kx) {
      int ih = h + ky - 1, iw = wq + kx - 1;
      if (ih >= 0 && ih < H && iw >= 0 && iw < W)
        acc += w[(ky * 3 + kx) * C + c] * xi[(long long)(ih * W + iw) * ldx + c];
    }
  u[(long long)l * C + c] = acc / (1.f + __expf(-acc));
}

// ------------------------ selective scan (Mamba) ----------------------------
// wave per channel; lanes = states; software-pipelined operand loads.
__global__ void scan_kernel(const float* __restrict__ u, const float* __restrict__ delta,
                            const float* __restrict__ A_log, const float* __restrict__ dbl,
                            int R2, int R, const float* __restrict__ D,
                            float* __restrict__ y, int L, int C, int rev)
{
  int wid  = threadIdx.x >> 5;
  int lane = threadIdx.x & 31;
  int c = blockIdx.x * (blockDim.x >> 5) + wid;
  if (c >= C) return;
  int n = lane & 15;
  float Acn = -__expf(A_log[(long long)c * NST + n]);
  float Dc  = D[c];
  float h = 0.f;
  int l = rev ? (L - 1) : 0;
  const int sd = rev ? -1 : 1;
  float dlt = delta[(long long)l * C + c];
  float uv  = u[(long long)l * C + c];
  float Bv  = dbl[(long long)l * R2 + R + n];
  float Cv  = dbl[(long long)l * R2 + R + NST + n];
  for (int step = 0; step < L; ++step) {
    int ln = l + sd;
    int lc = imin(imax(ln, 0), L - 1);
    float dltN = delta[(long long)lc * C + c];
    float uvN  = u[(long long)lc * C + c];
    float BvN  = dbl[(long long)lc * R2 + R + n];
    float CvN  = dbl[(long long)lc * R2 + R + NST + n];
    h = h * __expf(dlt * Acn) + dlt * uv * Bv;
    float contrib = h * Cv;
#pragma unroll
    for (int o = 8; o > 0; o >>= 1) contrib += __shfl_xor(contrib, o, 16);
    if (lane == 0) y[(long long)l * C + c] = contrib + uv * Dc;
    l = ln; dlt = dltN; uv = uvN; Bv = BvN; Cv = CvN;
  }
}

// ----------------------------- small elementwise ----------------------------
__global__ void mul_silu_kernel(const float* __restrict__ a, const float* __restrict__ hw2,
                                int C, float* __restrict__ t, long long total)
{
  long long i = blockIdx.x * (long long)blockDim.x + threadIdx.x;
  if (i >= total) return;
  int c = (int)(i % C);
  long long l = i / C;
  float z = hw2[l * (2LL * C) + C + c];
  t[i] = a[i] * (z / (1.f + __expf(-z)));
}

__global__ void add_kernel(float* __restrict__ x, const float* __restrict__ d, long long n)
{
  long long i = blockIdx.x * (long long)blockDim.x + threadIdx.x;
  if (i < n) x[i] += d[i];
}

__global__ void nchw_to_lc_kernel(const float* __restrict__ in, float* __restrict__ out,
                                  int C, int HW)
{
  long long i = blockIdx.x * (long long)blockDim.x + threadIdx.x;
  if (i >= (long long)C * HW) return;
  int c = (int)(i / HW), l = (int)(i % HW);
  out[(long long)l * C + c] = in[i];
}

__global__ void lc_to_nchw_kernel(const float* __restrict__ in, float* __restrict__ out,
                                  int C, int HW)
{
  long long i = blockIdx.x * (long long)blockDim.x + threadIdx.x;
  if (i >= (long long)C * HW) return;
  int c = (int)(i / HW), l = (int)(i % HW);
  out[i] = in[(long long)l * C + c];
}

// align-corners bilinear upsample, NCHW, optional output channel offset
__global__ void upsample_kernel(const float* __restrict__ in, float* __restrict__ out,
                                int C, int H, int W, int s, int ch_off)
{
  int Ho = H * s, Wo = W * s;
  long long i = blockIdx.x * (long long)blockDim.x + threadIdx.x;
  if (i >= (long long)C * Ho * Wo) return;
  int ox = (int)(i % Wo);
  long long t = i / Wo;
  int oy = (int)(t % Ho);
  int c  = (int)(t / Ho);
  float fy = (Ho > 1) ? oy * (float)(H - 1) / (float)(Ho - 1) : 0.f;
  float fx = (Wo > 1) ? ox * (float)(W - 1) / (float)(Wo - 1) : 0.f;
  int y0 = (int)floorf(fy); int y1 = imin(y0 + 1, H - 1);
  int x0 = (int)floorf(fx); int x1 = imin(x0 + 1, W - 1);
  float wy = fy - y0, wx = fx - x0;
  const float* p = in + (long long)c * H * W;
  float top = p[y0 * W + x0] * (1.f - wx) + p[y0 * W + x1] * wx;
  float bot = p[y1 * W + x0] * (1.f - wx) + p[y1 * W + x1] * wx;
  out[((long long)(ch_off + c) * Ho + oy) * Wo + ox] = top * (1.f - wy) + bot * wy;
}

// ============================ host orchestration ============================

struct VSSP { const float *ln_g,*ln_b,*in_w,*dw_w,*dw_b,*xproj_w,*dt_w,*dt_b,*A_log,*D,*on_g,*on_b,*out_w; };
struct CBN  { const float *w,*s,*b; };
struct BranchP { VSSP ccw1, cw1, ccw2, cw2; CBN cA, cB, upc; const float *w2,*b2; };

static inline unsigned gblocks(long long n) { return (unsigned)((n + 255) / 256); }

static void launch_gemm(hipStream_t st, const float* A, long long sAm, long long sAk,
                        const float* B, long long sBk, long long sBn,
                        float* C, long long sCm, long long sCn,
                        int M, int N, int K,
                        const float* scale, const float* bias, int act)
{
  dim3 g((N + BN - 1) / BN, (M + BM - 1) / BM);
  gemm_wmma_kernel<<<g, GT, 0, st>>>(A, sAm, sAk, B, sBk, sBn, C, sCm, sCn,
                                     M, N, K, scale, bias, act);
}

static void launch_conv3(hipStream_t st, const float* in, const CBN& q, float* out,
                         int Cin, int Cout, int H, int W, int act)
{
  dim3 g((Cout + BN - 1) / BN, (H * W + BM - 1) / BM);
  conv3x3_wmma_kernel<<<g, GT, 0, st>>>(in, q.w, out, Cin, Cout, H, W, q.s, q.b, act);
}

// x: [L,C] token-major, updated in place (residual). tA..tF scratch.
static void run_vss(hipStream_t st, float* x, int L, int H, int W, int C,
                    const VSSP& p, int rev,
                    float* tA, float* tB, float* tC, float* tD, float* tE, float* tF)
{
  const int R = (C + 15) / 16, R2 = R + 2 * NST;
  const long long tot = (long long)L * C;
  // h = LN(x)
  ln_kernel<<<(L + 7) / 8, 256, 0, st>>>(x, tA, p.ln_g, p.ln_b, L, C);
  // hw2 = h @ in_w   [L,2C]
  launch_gemm(st, tA, C, 1, p.in_w, 2LL * C, 1, tB, 2LL * C, 1, L, 2 * C, C,
              nullptr, nullptr, 0);
  // u = silu(dwconv3x3(xi) + dw_b)   xi = hw2[:, :C]
  dwconv_silu_kernel<<<gblocks(tot), 256, 0, st>>>(tB, 2 * C, p.dw_w, p.dw_b, tC, H, W, C);
  // dbl = u @ xproj_w   [L, R+32]
  launch_gemm(st, tC, C, 1, p.xproj_w, R2, 1, tD, R2, 1, L, R2, C,
              nullptr, nullptr, 0);
  // delta = softplus(dtr @ dt_w + dt_b)   dtr = dbl[:, :R]
  launch_gemm(st, tD, R2, 1, p.dt_w, C, 1, tE, C, 1, L, C, R,
              nullptr, p.dt_b, 4);
  // y = selective_scan(u, delta, A, B, C, D)
  scan_kernel<<<(C + 3) / 4, 128, 0, st>>>(tC, tE, p.A_log, tD, R2, R, p.D, tF, L, C, rev);
  // y = LN(y)
  ln_kernel<<<(L + 7) / 8, 256, 0, st>>>(tF, tA, p.on_g, p.on_b, L, C);
  // t = y * silu(z)   z = hw2[:, C:]
  mul_silu_kernel<<<gblocks(tot), 256, 0, st>>>(tA, tB, C, tF, tot);
  // y2 = t @ out_w
  launch_gemm(st, tF, C, 1, p.out_w, C, 1, tC, C, 1, L, C, C, nullptr, nullptr, 0);
  // x += y2
  add_kernel<<<gblocks(tot), 256, 0, st>>>(x, tC, tot);
}

extern "C" void kernel_launch(void* const* d_in, const int* in_sizes, int n_in,
                              void* d_out, int out_size, void* d_ws, size_t ws_size,
                              hipStream_t stream)
{
  (void)in_sizes; (void)n_in; (void)out_size; (void)ws_size;

  // ---------------- parameter walk (setup_inputs dict insertion order) ------
  int cur = 0;
  auto next = [&]() { return (const float*)d_in[cur++]; };
  const float* x2 = next();                 // [1,128,64,64]
  const float* x1 = next();                 // [1,512,16,16]
  auto getCBN = [&]() { CBN q; q.w = next(); q.s = next(); q.b = next(); return q; };
  auto getVSS = [&]() {
    VSSP v; v.ln_g = next(); v.ln_b = next(); v.in_w = next(); v.dw_w = next();
    v.dw_b = next(); v.xproj_w = next(); v.dt_w = next(); v.dt_b = next();
    v.A_log = next(); v.D = next(); v.on_g = next(); v.on_b = next();
    v.out_w = next(); return v;
  };
  CBN conv_x1 = getCBN();
  CBN conv_x2 = getCBN();
  BranchP br[2];
  for (int i = 0; i < 2; ++i) {
    br[i].ccw1 = getVSS(); br[i].cw1 = getVSS();
    br[i].ccw2 = getVSS(); br[i].cw2 = getVSS();
    br[i].cA = getCBN(); br[i].cB = getCBN();
    br[i].upc = getCBN(); br[i].w2 = next(); br[i].b2 = next();
  }
  VSSP mccw = getVSS(), mcw = getVSS();
  CBN mcA = getCBN(), mcB = getCBN(), mupc = getCBN();
  const float* mw2 = next(); const float* mb2 = next();

  // ---------------- workspace arena (floats) -------------------------------
  float* W0 = (float*)d_ws;
  float* P0  = W0;                    //  8,388,608  (512x128x128 up2 buffer)
  float* P1  = W0 + 8388608LL;        //  5,242,880  (hw2 @640 / 256x128x128)
  float* P2  = W0 + 13631488LL;       //  2,752,512
  float* P3  = W0 + 16384000LL;       //  2,752,512
  float* P4  = W0 + 19136512LL;       //  2,752,512
  float* P5  = W0 + 21889024LL;       //  2,752,512
  float* P6  = W0 + 24641536LL;       //  2,752,512
  float* P7  = W0 + 27394048LL;       //  2,752,512
  float* P8  = W0 + 30146560LL;       //  1,048,576  (x1h lc)
  float* P9  = W0 + 31195136LL;       //  1,048,576  (x1h nchw)
  float* P10 = W0 + 32243712LL;       //  1,048,576  (x2h)
  // total ~33.3M floats (~127 MB)

  float* out0 = (float*)d_out;
  const long long OUT_SEG = 256LL * 128 * 128;

  // ---------------- common "tail": convs + up2 + final 1x1 -----------------
  auto tail = [&](const CBN& cA, const CBN& cB, const CBN& upc,
                  const float* w2, const float* b2, float* out) {
    // input: [640,64,64] NCHW at P2
    launch_conv3(stream, P2, cA, P4, 640, 512, 64, 64, 1);
    launch_conv3(stream, P4, cB, P5, 512, 512, 64, 64, 1);
    upsample_kernel<<<gblocks(512LL * 128 * 128), 256, 0, stream>>>(P5, P0, 512, 64, 64, 2, 0);
    launch_conv3(stream, P0, upc, P1, 512, 256, 128, 128, 1);
    // 1x1 conv: out[o,hw] = sum_c w2[o,c]*x[c,hw] + b2[o]
    launch_gemm(stream, P1, 1, 128LL * 128, w2, 1, 256, out, 1, 128LL * 128,
                128 * 128, 256, 256, nullptr, b2, 0);
  };

  // ---------------- branch (back / fore) -----------------------------------
  auto do_branch = [&](const float* x1h /*[512,16,16]*/, const float* x2h /*[128,64,64]*/,
                       const BranchP& bp, float* out) {
    nchw_to_lc_kernel<<<gblocks(512LL * 256), 256, 0, stream>>>(x1h, P8, 512, 256);
    run_vss(stream, P8, 256, 16, 16, 512, bp.ccw1, 0, P2, P3, P4, P5, P6, P7);
    run_vss(stream, P8, 256, 16, 16, 512, bp.cw1, 1, P2, P3, P4, P5, P6, P7);
    lc_to_nchw_kernel<<<gblocks(512LL * 256), 256, 0, stream>>>(P8, P9, 512, 256);
    // concat([x2h, up4(x1h')]) -> [640,64,64] at P2
    hipMemcpyAsync(P2, x2h, 128LL * 4096 * sizeof(float), hipMemcpyDeviceToDevice, stream);
    upsample_kernel<<<gblocks(512LL * 64 * 64), 256, 0, stream>>>(P9, P2, 512, 16, 16, 4, 128);
    nchw_to_lc_kernel<<<gblocks(640LL * 4096), 256, 0, stream>>>(P2, P3, 640, 4096);
    run_vss(stream, P3, 4096, 64, 64, 640, bp.ccw2, 0, P4, P1, P5, P6, P7, P2);
    run_vss(stream, P3, 4096, 64, 64, 640, bp.cw2, 1, P4, P1, P5, P6, P7, P2);
    lc_to_nchw_kernel<<<gblocks(640LL * 4096), 256, 0, stream>>>(P3, P2, 640, 4096);
    tail(bp.cA, bp.cB, bp.upc, bp.w2, bp.b2, out);
  };

  // ---------------- back branch --------------------------------------------
  launch_gemm(stream, x2, 1, 4096, conv_x2.w, 1, 64, P10, 1, 4096,
              4096, 128, 64, conv_x2.s, conv_x2.b, 2);
  launch_gemm(stream, x1, 1, 256, conv_x1.w, 1, 256, P9, 1, 256,
              256, 512, 256, conv_x1.s, conv_x1.b, 2);
  do_branch(P9, P10, br[0], out0 + OUT_SEG);

  // ---------------- fore branch --------------------------------------------
  launch_gemm(stream, x2 + 64LL * 4096, 1, 4096, conv_x2.w, 1, 64, P10, 1, 4096,
              4096, 128, 64, conv_x2.s, conv_x2.b, 2);
  launch_gemm(stream, x1 + 256LL * 256, 1, 256, conv_x1.w, 1, 256, P9, 1, 256,
              256, 512, 256, conv_x1.s, conv_x1.b, 2);
  do_branch(P9, P10, br[1], out0 + 2 * OUT_SEG);

  // ---------------- merge / main path --------------------------------------
  nchw_to_lc_kernel<<<gblocks(512LL * 256), 256, 0, stream>>>(x1, P8, 512, 256);
  run_vss(stream, P8, 256, 16, 16, 512, mccw, 0, P2, P3, P4, P5, P6, P7);
  run_vss(stream, P8, 256, 16, 16, 512, mcw, 1, P2, P3, P4, P5, P6, P7);
  lc_to_nchw_kernel<<<gblocks(512LL * 256), 256, 0, stream>>>(P8, P9, 512, 256);
  hipMemcpyAsync(P2, x2, 128LL * 4096 * sizeof(float), hipMemcpyDeviceToDevice, stream);
  upsample_kernel<<<gblocks(512LL * 64 * 64), 256, 0, stream>>>(P9, P2, 512, 16, 16, 4, 128);
  tail(mcA, mcB, mupc, mw2, mb2, out0);
}
